// StripedAttention_65687229825463
// MI455X (gfx1250) — matrix operand order repaired and checked by reading, not compiled, not run on hardware
//
#include <hip/hip_runtime.h>
#include <stdint.h>

#define D_MODEL 2048
#define N_HEADS 16
#define HEAD_DIM 128
#define WINDOW 256
#define BATCH 2
#define SEQ 2048
#define MTOT (BATCH * SEQ)

typedef __attribute__((ext_vector_type(16))) __bf16   v16bf;
typedef __attribute__((ext_vector_type(8)))  float    v8f;
typedef __attribute__((ext_vector_type(4)))  uint32_t v4u;

union FragU { v16bf v; v4u u[2]; };

__device__ __forceinline__ uint16_t f2bf(float x) {
  uint32_t u = __float_as_uint(x);
  return (uint16_t)((u + 0x7FFFu + ((u >> 16) & 1u)) >> 16);   // round-nearest-even
}

// A-operand fragment (16x32 bf16): lane m holds K = g*8..g*8+7 and 16+g*8..+7
__device__ __forceinline__ v16bf load_a_frag(const uint16_t* p) {
  FragU f;
  f.u[0] = *reinterpret_cast<const v4u*>(p);
  f.u[1] = *reinterpret_cast<const v4u*>(p + 16);
  return f.v;
}
// B-operand fragment (32x16 bf16): lane n holds column n, K = g*8..g*8+7 and 16+g*8..+7
__device__ __forceinline__ v16bf load_b_frag(const uint16_t* p) {
  FragU f;
  f.u[0] = *reinterpret_cast<const v4u*>(p);
  f.u[1] = *reinterpret_cast<const v4u*>(p + 16);
  return f.v;
}

__device__ __forceinline__ v8f wmma_bf16(v16bf a, v16bf b, v8f c) {
  return __builtin_amdgcn_wmma_f32_16x16x32_bf16(false, a, false, b, (short)0, c, false, false);
}

// ---- CDNA5 async global->LDS copy (ASYNCcnt path), inline asm = toolchain-portable ----
__device__ __forceinline__ void async_copy_b128(uint32_t lds_off, const uint16_t* g) {
  asm volatile("global_load_async_to_lds_b128 %0, %1, off"
               :: "v"(lds_off), "v"(g) : "memory");
}
__device__ __forceinline__ void wait_asynccnt_le5() {
  asm volatile("s_wait_asynccnt 5" ::: "memory");
}
__device__ __forceinline__ void wait_asynccnt_0() {
  asm volatile("s_wait_asynccnt 0" ::: "memory");
}

// ---------------- fp32 -> bf16 conversion ----------------
__global__ __launch_bounds__(256) void cvt_bf16_kernel(const float* __restrict__ in,
                                                       uint16_t* __restrict__ out, int n) {
  int i = blockIdx.x * blockDim.x + threadIdx.x;
  int stride = gridDim.x * blockDim.x;
  for (; i < n; i += stride) out[i] = f2bf(in[i]);
}

// ---------------- bf16 GEMM: C[m,n] = sum_k A[m,k] * Bw[n,k] ----------------
// Block tile 64(M) x 256(N), 8 waves, each wave 32x64. K staged 32 at a time through
// double-buffered LDS filled with global_load_async_to_lds_b128.
// MODE 0: bf16 -> [B][H][T][128]   (Q / K)
// MODE 1: bf16 -> [B][H][128][T]   (V transposed; packed b128 stores)
// MODE 2: fp32 row-major [M][N]    (final output)
#define APITCH 40   // ushorts per 32-elem row (+16B pad => 20-dword pitch, conflict-free)
template <int MODE>
__global__ __launch_bounds__(256) void gemm_bf16_kernel(const uint16_t* __restrict__ A,
                                                        const uint16_t* __restrict__ Bw,
                                                        void* __restrict__ Cout) {
  constexpr int K = D_MODEL;
  constexpr int NK = K / 32;
  __shared__ __attribute__((aligned(32))) uint16_t As[2][64 * APITCH];
  __shared__ __attribute__((aligned(32))) uint16_t Bs[2][256 * APITCH];

  const int tid  = threadIdx.x;
  const int lane = tid & 31;
  const int wave = tid >> 5;
  const int g    = lane >> 4;
  const int hl   = lane & 15;
  const int waveM = wave >> 2;   // 0..1
  const int waveN = wave & 3;    // 0..3
  const int blkM = blockIdx.x * 64;
  const int blkN = blockIdx.y * 256;

  // staging assignment: thread -> (row = tid>>2, 16B chunk c = tid&3)
  const int srow = tid >> 2;
  const int sc   = tid & 3;
  const uint16_t* gA = A  + (size_t)(blkM + srow) * K + sc * 8;
  const uint16_t* gB0 = Bw + (size_t)(blkN + srow) * K + sc * 8;

  v8f acc[2][4];
#pragma unroll
  for (int i = 0; i < 2; ++i)
#pragma unroll
    for (int j = 0; j < 4; ++j) { v8f z = {}; acc[i][j] = z; }

  auto issue = [&](int buf, int k0) {
    async_copy_b128((uint32_t)(uintptr_t)&As[buf][srow * APITCH + sc * 8], gA + k0);
#pragma unroll
    for (int j = 0; j < 4; ++j)
      async_copy_b128((uint32_t)(uintptr_t)&Bs[buf][(srow + 64 * j) * APITCH + sc * 8],
                      gB0 + (size_t)(64 * j) * K + k0);
  };

  issue(0, 0);
  for (int ks = 0; ks < NK; ++ks) {
    const int cur = ks & 1;
    const bool more = (ks + 1 < NK);
    if (more) issue(1 - cur, (ks + 1) * 32);
    if (more) wait_asynccnt_le5(); else wait_asynccnt_0();
    __syncthreads();

    v16bf af[2], bf[4];
#pragma unroll
    for (int i = 0; i < 2; ++i)
      af[i] = load_a_frag(&As[cur][(waveM * 32 + i * 16 + hl) * APITCH + g * 8]);
#pragma unroll
    for (int j = 0; j < 4; ++j)
      bf[j] = load_b_frag(&Bs[cur][(waveN * 64 + j * 16 + hl) * APITCH + g * 8]);
#pragma unroll
    for (int i = 0; i < 2; ++i)
#pragma unroll
      for (int j = 0; j < 4; ++j)
        acc[i][j] = wmma_bf16(af[i], bf[j], acc[i][j]);

    __syncthreads();   // all reads of buf[cur] done before it is refilled at ks+2
  }

  const int tileM = blkM + waveM * 32;
  const int tileN = blkN + waveN * 64;
#pragma unroll
  for (int i = 0; i < 2; ++i) {
#pragma unroll
    for (int j = 0; j < 4; ++j) {
      const int mbase = tileM + i * 16 + g * 8;   // C layout: VGPR r -> row mbase+r
      const int n     = tileN + j * 16 + hl;      // lane column
      if (MODE == 2) {
        float* C = (float*)Cout;
#pragma unroll
        for (int r = 0; r < 8; ++r)
          C[(size_t)(mbase + r) * D_MODEL + n] = acc[i][j][r];
      } else if (MODE == 0) {
        uint16_t* C = (uint16_t*)Cout;
        const int h = n >> 7, d = n & (HEAD_DIM - 1);
#pragma unroll
        for (int r = 0; r < 8; ++r) {
          const int m = mbase + r;
          const int b = m >> 11, t = m & (SEQ - 1);
          C[(size_t)((b * N_HEADS + h) * SEQ + t) * HEAD_DIM + d] = f2bf(acc[i][j][r]);
        }
      } else {  // MODE 1: V^T, 8 consecutive t per lane -> one packed 16B store
        uint16_t* C = (uint16_t*)Cout;
        const int h = n >> 7, d = n & (HEAD_DIM - 1);
        const int b = mbase >> 11, t = mbase & (SEQ - 1);
        uint32_t w0 = (uint32_t)f2bf(acc[i][j][0]) | ((uint32_t)f2bf(acc[i][j][1]) << 16);
        uint32_t w1 = (uint32_t)f2bf(acc[i][j][2]) | ((uint32_t)f2bf(acc[i][j][3]) << 16);
        uint32_t w2 = (uint32_t)f2bf(acc[i][j][4]) | ((uint32_t)f2bf(acc[i][j][5]) << 16);
        uint32_t w3 = (uint32_t)f2bf(acc[i][j][6]) | ((uint32_t)f2bf(acc[i][j][7]) << 16);
        v4u pk = {w0, w1, w2, w3};
        *reinterpret_cast<v4u*>(&C[(size_t)((b * N_HEADS + h) * HEAD_DIM + d) * SEQ + t]) = pk;
      }
    }
  }
}

// ---------------- flash attention: one wave per (b, h, 16-query tile) ----------------
__global__ __launch_bounds__(128) void attn_kernel(const uint16_t* __restrict__ Qb,
                                                   const uint16_t* __restrict__ Kb,
                                                   const uint16_t* __restrict__ Vt,
                                                   uint16_t* __restrict__ Ob) {
  __shared__ __attribute__((aligned(32))) uint16_t smem[4][16 * 32];  // per-wave P tile
  const int lane = threadIdx.x & 31;
  const int wave = threadIdx.x >> 5;
  const int g    = lane >> 4;
  const int hl   = lane & 15;
  const int qt = blockIdx.x * 4 + wave;
  const int h  = blockIdx.y;
  const int b  = blockIdx.z;
  const int t0 = qt * 16;
  const bool is_global = (h & 1) == 0;
  const float CSCALE = 0.12751542f;  // log2(e) / sqrt(128)

  const uint16_t* Qp = Qb + ((size_t)(b * N_HEADS + h) * SEQ + t0) * HEAD_DIM;
  const uint16_t* Kp = Kb + ((size_t)(b * N_HEADS + h) * SEQ) * HEAD_DIM;
  const uint16_t* Vp = Vt + ((size_t)(b * N_HEADS + h) * HEAD_DIM) * SEQ;

  v16bf qfrag[4];
#pragma unroll
  for (int kk = 0; kk < 4; ++kk)
    qfrag[kk] = load_a_frag(Qp + (size_t)hl * HEAD_DIM + kk * 32 + g * 8);

  v8f O[8];
  float mrow[8], lrow[8];
#pragma unroll
  for (int dt = 0; dt < 8; ++dt) { v8f z = {}; O[dt] = z; }
#pragma unroll
  for (int r = 0; r < 8; ++r) { mrow[r] = -1.0e30f; lrow[r] = 0.0f; }

  int kstart = 0;
  if (!is_global) {
    int ks = t0 - (WINDOW - 1);
    if (ks < 0) ks = 0;
    kstart = ks & ~31;
  }

  for (int kb = kstart; kb <= t0 + 15; kb += 32) {
    // S = Q K^T for 32 keys (two 16-key WMMA column tiles)
    v8f S0 = {}, S1 = {};
#pragma unroll
    for (int kk = 0; kk < 4; ++kk) {
      v16bf bk0 = load_b_frag(Kp + (size_t)(kb + hl) * HEAD_DIM + kk * 32 + g * 8);
      v16bf bk1 = load_b_frag(Kp + (size_t)(kb + 16 + hl) * HEAD_DIM + kk * 32 + g * 8);
      S0 = wmma_bf16(qfrag[kk], bk0, S0);
      S1 = wmma_bf16(qfrag[kk], bk1, S1);
    }
    const int rowb = t0 + 8 * g;
    const int key0 = kb + hl;
    const int key1 = key0 + 16;
#pragma unroll
    for (int r = 0; r < 8; ++r) {
      const int row = rowb + r;
      float s0 = S0[r], s1 = S1[r];
      const bool a0 = (key0 <= row) && (is_global || key0 > row - WINDOW);
      const bool a1 = (key1 <= row) && (is_global || key1 > row - WINDOW);
      s0 = a0 ? s0 : -3.0e38f;
      s1 = a1 ? s1 : -3.0e38f;
      // row max across the 16 key lanes (stays inside each 16-lane half = this row group)
      float mx = fmaxf(s0, s1);
#pragma unroll
      for (int msk = 1; msk < 16; msk <<= 1) mx = fmaxf(mx, __shfl_xor(mx, msk, 32));
      const float mnew = fmaxf(mrow[r], mx);
      const float corr = exp2f((mrow[r] - mnew) * CSCALE);
      const float e0 = exp2f((s0 - mnew) * CSCALE);
      const float e1 = exp2f((s1 - mnew) * CSCALE);
      float sum = e0 + e1;
#pragma unroll
      for (int msk = 1; msk < 16; msk <<= 1) sum += __shfl_xor(sum, msk, 32);
      lrow[r] = lrow[r] * corr + sum;
      mrow[r] = mnew;
#pragma unroll
      for (int dt = 0; dt < 8; ++dt) O[dt][r] *= corr;
      // stage P (C layout) into LDS for re-read in A layout; same-wave LDS is in-order
      smem[wave][(8 * g + r) * 32 + hl]      = f2bf(e0);
      smem[wave][(8 * g + r) * 32 + 16 + hl] = f2bf(e1);
    }
    v16bf pf = load_a_frag(&smem[wave][hl * 32 + g * 8]);  // P as 16x32 A operand
#pragma unroll
    for (int dt = 0; dt < 8; ++dt) {
      v16bf vf = load_b_frag(Vp + (size_t)(dt * 16 + hl) * SEQ + kb + g * 8);
      O[dt] = wmma_bf16(pf, vf, O[dt]);
    }
  }

  // normalize and store in concat layout [B][T][H*128] (bf16)
#pragma unroll
  for (int dt = 0; dt < 8; ++dt) {
#pragma unroll
    for (int r = 0; r < 8; ++r) {
      const int row = t0 + 8 * g + r;
      const float val = O[dt][r] / lrow[r];
      Ob[(size_t)(b * SEQ + row) * D_MODEL + h * HEAD_DIM + dt * 16 + hl] = f2bf(val);
    }
  }
}

extern "C" void kernel_launch(void* const* d_in, const int* in_sizes, int n_in,
                              void* d_out, int out_size, void* d_ws, size_t ws_size,
                              hipStream_t stream) {
  (void)in_sizes; (void)n_in; (void)out_size; (void)ws_size;
  const float* x  = (const float*)d_in[0];
  const float* Wq = (const float*)d_in[1];
  const float* Wk = (const float*)d_in[2];
  const float* Wv = (const float*)d_in[3];
  const float* Wo = (const float*)d_in[4];

  uint8_t* ws = (uint8_t*)d_ws;
  size_t off = 0;
  auto take = [&](size_t elems) {
    uint16_t* p = (uint16_t*)(ws + off);
    off += ((elems * sizeof(uint16_t)) + 255) & ~(size_t)255;
    return p;
  };
  uint16_t* xb  = take((size_t)MTOT * D_MODEL);
  uint16_t* Wqb = take((size_t)D_MODEL * D_MODEL);
  uint16_t* Wkb = take((size_t)D_MODEL * D_MODEL);
  uint16_t* Wvb = take((size_t)D_MODEL * D_MODEL);
  uint16_t* Wob = take((size_t)D_MODEL * D_MODEL);
  uint16_t* Qb  = take((size_t)MTOT * D_MODEL);
  uint16_t* Kbf = take((size_t)MTOT * D_MODEL);
  uint16_t* Vt  = take((size_t)MTOT * D_MODEL);
  uint16_t* Ab  = take((size_t)MTOT * D_MODEL);

  cvt_bf16_kernel<<<4096, 256, 0, stream>>>(x,  xb,  MTOT * D_MODEL);
  cvt_bf16_kernel<<<2048, 256, 0, stream>>>(Wq, Wqb, D_MODEL * D_MODEL);
  cvt_bf16_kernel<<<2048, 256, 0, stream>>>(Wk, Wkb, D_MODEL * D_MODEL);
  cvt_bf16_kernel<<<2048, 256, 0, stream>>>(Wv, Wvb, D_MODEL * D_MODEL);
  cvt_bf16_kernel<<<2048, 256, 0, stream>>>(Wo, Wob, D_MODEL * D_MODEL);

  dim3 gg(MTOT / 64, D_MODEL / 256);
  gemm_bf16_kernel<0><<<gg, 256, 0, stream>>>(xb, Wqb, Qb);
  gemm_bf16_kernel<0><<<gg, 256, 0, stream>>>(xb, Wkb, Kbf);
  gemm_bf16_kernel<1><<<gg, 256, 0, stream>>>(xb, Wvb, Vt);

  attn_kernel<<<dim3(SEQ / 16 / 4, N_HEADS, BATCH), 128, 0, stream>>>(Qb, Kbf, Vt, Ab);

  gemm_bf16_kernel<2><<<gg, 256, 0, stream>>>(Ab, Wob, (float*)d_out);
}
